// Seq2Seq_3169685865362
// MI455X (gfx1250) — matrix operand order repaired and checked by reading, not compiled, not run on hardware
//
#include <hip/hip_runtime.h>
#include <hip/hip_bf16.h>

// ---- problem constants (match reference) ----
#define LNUM 4
#define HSZ  1024
#define ESZ  1024
#define HDIR 512          // per-direction encoder hidden
#define VOUTN 32000
#define BSZ  16
#define SLEN 50
#define TLEN 50
#define NB   4            // N-tiles per wave (register blocking); all N % 64 == 0

typedef __attribute__((ext_vector_type(16))) _Float16 v16h;
typedef __attribute__((ext_vector_type(8)))  float    v8f;

__device__ __forceinline__ float sigmoidf_(float x) { return 1.0f / (1.0f + __expf(-x)); }

// B fragment: 16 contiguous fp32 -> f16 (lane&15 = N col; lanes>=16 K base +16)
__device__ __forceinline__ v16h load_cvt_b16(const float* __restrict__ p) {
  v16h r;
#pragma unroll
  for (int j = 0; j < 16; ++j) r[j] = (_Float16)p[j];
  return r;
}

// ---------------------------------------------------------------------------
// 16x64 WMMA tile-row accumulation: acc[nb] += A_tile[16,K] * W_tile_nb[16,K]^T
// A fragment loaded/converted once per k-step, reused across 4 WMMAs.
// The four B fragments are distinct named values so their 128-bit global
// loads issue independently; per-wave latency gaps are covered by occupancy
// (moderate VGPR footprint -> 8 waves/SIMD).
// Operand k order used here (f16, 16x16x32; contiguous k, same for A and B, not the ISA table order):
//   A (16x32): lane&15 = M row; lanes>=16 K base +16; elem j -> K=base+j
//   B (32x16): lane&15 = N col; lanes>=16 K base +16; elem j -> K=base+j
// ---------------------------------------------------------------------------
__device__ __forceinline__ void wmma_tile4_f32f16(const float* __restrict__ Atile, int lda,
                                                  const float* __restrict__ Wtile, int ldw,
                                                  int K, int lane, v8f acc[NB]) {
  const float* arow  = Atile + (size_t)(lane & 15) * lda + ((lane >= 16) ? 16 : 0);
  const float* wrow0 = Wtile + (size_t)(lane & 15) * ldw + ((lane >= 16) ? 16 : 0);
  const float* wrow1 = wrow0 + (size_t)16 * ldw;
  const float* wrow2 = wrow0 + (size_t)32 * ldw;
  const float* wrow3 = wrow0 + (size_t)48 * ldw;
#pragma unroll 2
  for (int k0 = 0; k0 < K; k0 += 32) {
    v16h a;
#pragma unroll
    for (int j = 0; j < 8; ++j) {
      a[j]     = (_Float16)arow[k0 + j];
      a[j + 8] = (_Float16)arow[k0 + 8 + j];
    }
    // keep the weight stream ahead of the cvt/WMMA pipeline
    __builtin_prefetch((const void*)(wrow0 + k0 + 64), 0, 1);
    v16h b0 = load_cvt_b16(wrow0 + k0);
    v16h b1 = load_cvt_b16(wrow1 + k0);
    v16h b2 = load_cvt_b16(wrow2 + k0);
    v16h b3 = load_cvt_b16(wrow3 + k0);
    acc[0] = __builtin_amdgcn_wmma_f32_16x16x32_f16(false, a, false, b0, (short)0,
                                                    acc[0], false, false);
    acc[1] = __builtin_amdgcn_wmma_f32_16x16x32_f16(false, a, false, b1, (short)0,
                                                    acc[1], false, false);
    acc[2] = __builtin_amdgcn_wmma_f32_16x16x32_f16(false, a, false, b2, (short)0,
                                                    acc[2], false, false);
    acc[3] = __builtin_amdgcn_wmma_f32_16x16x32_f16(false, a, false, b3, (short)0,
                                                    acc[3], false, false);
  }
}

__device__ __forceinline__ void zero_acc4(v8f acc[NB]) {
#pragma unroll
  for (int nb = 0; nb < NB; ++nb) acc[nb] = (v8f){0.f, 0.f, 0.f, 0.f, 0.f, 0.f, 0.f, 0.f};
}

// ---------------------------------------------------------------------------
// Generic GEMM: C[M,N] = A[M,K] * W[N,K]^T (+bias1 +bias2), fp32 row-major.
// One wave per 16x64 output tile-row (4 WMMA tiles).
// ---------------------------------------------------------------------------
__global__ void gemm_wmma_bias_kernel(const float* __restrict__ A, int lda,
                                      const float* __restrict__ W, int ldw,
                                      const float* __restrict__ bias1,
                                      const float* __restrict__ bias2,
                                      float* __restrict__ C, int ldc,
                                      int M, int N, int K) {
  int gwave = (blockIdx.x * blockDim.x + threadIdx.x) >> 5;
  int lane  = threadIdx.x & 31;
  int ngrp  = N >> 6;                        // groups of 64 columns
  int total = (M >> 4) * ngrp;
  if (gwave >= total) return;                // wave-uniform exit
  int tm = gwave / ngrp, tg = gwave - tm * ngrp;
  v8f acc[NB];
  zero_acc4(acc);
  wmma_tile4_f32f16(A + (size_t)tm * 16 * lda, lda,
                    W + (size_t)tg * 64 * ldw, ldw, K, lane, acc);
  int r0 = tm * 16 + ((lane >= 16) ? 8 : 0);
#pragma unroll
  for (int nb = 0; nb < NB; ++nb) {
    int col = tg * 64 + nb * 16 + (lane & 15);
    float badd = (bias1 ? bias1[col] : 0.f) + (bias2 ? bias2[col] : 0.f);
#pragma unroll
    for (int i = 0; i < 8; ++i)
      C[(size_t)(r0 + i) * ldc + col] = acc[nb][i] + badd;
  }
}

// ---------------------------------------------------------------------------
// Embedding gather: out[s][b][:] = table[idx[b][s]][:]   (fp32)
// ---------------------------------------------------------------------------
__global__ void gather_embed_kernel(const int* __restrict__ idx,
                                    const float* __restrict__ table,
                                    float* __restrict__ out, int Bv, int Sv, int Ev) {
  int sb = blockIdx.x;           // s*B + b
  int s = sb / Bv, b = sb - s * Bv;
  int row = idx[b * Sv + s];
  const float* src = table + (size_t)row * Ev;
  float* dst = out + (size_t)sb * Ev;
  for (int e = threadIdx.x; e < Ev; e += blockDim.x) dst[e] = src[e];
}

// ---------------------------------------------------------------------------
// Encoder recurrent step GEMM: g[dir] = xg[dir][t] + h_cur[dir] @ Whh[dir]^T
// blockIdx.z = direction.  M=16 (batch), N=4*HDIR=2048, K=HDIR=512.
// ---------------------------------------------------------------------------
__global__ void enc_step_gemm_kernel(const float* __restrict__ h_cur,  // [2][B][HDIR]
                                     const float* __restrict__ Whh,    // [2][4HDIR][HDIR]
                                     const float* __restrict__ xg,     // [2][S][B][4HDIR]
                                     float* __restrict__ g_out,        // [2][B][4HDIR]
                                     int st) {
  int dir   = blockIdx.z;
  int gwave = (blockIdx.x * blockDim.x + threadIdx.x) >> 5;
  int lane  = threadIdx.x & 31;
  const int N = 4 * HDIR;
  if (gwave >= (N >> 6)) return;
  int tg = gwave;
  const float* A = h_cur + (size_t)dir * BSZ * HDIR;
  const float* W = Whh + (size_t)dir * N * HDIR + (size_t)tg * 64 * HDIR;
  v8f acc[NB];
  zero_acc4(acc);
  wmma_tile4_f32f16(A, HDIR, W, HDIR, HDIR, lane, acc);
  int t = dir ? (SLEN - 1 - st) : st;
  const float* xg_t = xg + (size_t)dir * SLEN * BSZ * N + (size_t)t * BSZ * N;
  float* go = g_out + (size_t)dir * BSZ * N;
  int r0 = (lane >= 16) ? 8 : 0;
#pragma unroll
  for (int nb = 0; nb < NB; ++nb) {
    int col = tg * 64 + nb * 16 + (lane & 15);
#pragma unroll
    for (int i = 0; i < 8; ++i) {
      int r = r0 + i;
      go[(size_t)r * N + col] = acc[nb][i] + xg_t[(size_t)r * N + col];
    }
  }
}

// ---------------------------------------------------------------------------
// Encoder LSTM cell (per direction), writes ys[t][b][dir*HDIR+j] and, at the
// last step, the decoder's initial h0/c0 (concat fwd/bwd) for this layer.
// ---------------------------------------------------------------------------
__global__ void enc_cell_kernel(const float* __restrict__ g_all,  // [2][B][4HDIR]
                                float* __restrict__ h_cur, float* __restrict__ c_cur,
                                float* __restrict__ ys,           // [S][B][H]
                                int st, float* __restrict__ dec_h,
                                float* __restrict__ dec_c, int layer) {
  int dir = blockIdx.z;
  int idx = blockIdx.x * blockDim.x + threadIdx.x;
  if (idx >= BSZ * HDIR) return;
  int b = idx / HDIR, j = idx - b * HDIR;
  const float* g = g_all + (size_t)dir * BSZ * 4 * HDIR + (size_t)b * 4 * HDIR;
  float* c = c_cur + (size_t)dir * BSZ * HDIR;
  float* h = h_cur + (size_t)dir * BSZ * HDIR;
  float iv = sigmoidf_(g[j]);
  float fv = sigmoidf_(g[HDIR + j]);
  float gv = tanhf(g[2 * HDIR + j]);
  float ov = sigmoidf_(g[3 * HDIR + j]);
  float cc = fv * c[idx] + iv * gv;
  float hh = ov * tanhf(cc);
  c[idx] = cc;
  h[idx] = hh;
  int t = dir ? (SLEN - 1 - st) : st;
  ys[((size_t)t * BSZ + b) * HSZ + dir * HDIR + j] = hh;
  if (st == SLEN - 1) {
    dec_h[((size_t)layer * BSZ + b) * HSZ + dir * HDIR + j] = hh;
    dec_c[((size_t)layer * BSZ + b) * HSZ + dir * HDIR + j] = cc;
  }
}

// ---------------------------------------------------------------------------
// Decoder gate GEMM: g[B,4H] = x0@Wih[:, :K0]^T (+ x1@Wih[:, K0:]^T)
//                              + h@Whh^T + bih + bhh
// (x1 carries the input-feeding h_tilde for layer 0; null for layers 1..3)
// ---------------------------------------------------------------------------
__global__ void dec_gates_kernel(const float* __restrict__ x0, int K0,
                                 const float* __restrict__ x1, int K1,
                                 const float* __restrict__ Wih, int ldwih,
                                 const float* __restrict__ h,
                                 const float* __restrict__ Whh,
                                 const float* __restrict__ bih,
                                 const float* __restrict__ bhh,
                                 float* __restrict__ g_out) {
  int gwave = (blockIdx.x * blockDim.x + threadIdx.x) >> 5;
  int lane  = threadIdx.x & 31;
  const int N = 4 * HSZ;
  if (gwave >= (N >> 6)) return;
  int tg = gwave;
  v8f acc[NB];
  zero_acc4(acc);
  wmma_tile4_f32f16(x0, K0, Wih + (size_t)tg * 64 * ldwih, ldwih, K0, lane, acc);
  if (x1)
    wmma_tile4_f32f16(x1, K1, Wih + (size_t)tg * 64 * ldwih + K0, ldwih, K1, lane, acc);
  wmma_tile4_f32f16(h, HSZ, Whh + (size_t)tg * 64 * HSZ, HSZ, HSZ, lane, acc);
  int r0 = (lane >= 16) ? 8 : 0;
#pragma unroll
  for (int nb = 0; nb < NB; ++nb) {
    int col = tg * 64 + nb * 16 + (lane & 15);
    float badd = bih[col] + bhh[col];
#pragma unroll
    for (int i = 0; i < 8; ++i)
      g_out[(size_t)(r0 + i) * N + col] = acc[nb][i] + badd;
  }
}

// ---------------------------------------------------------------------------
// Decoder LSTM cell: in-place update of h[l], c[l] from gates g.
// ---------------------------------------------------------------------------
__global__ void dec_cell_kernel(const float* __restrict__ g,
                                float* __restrict__ h, float* __restrict__ c) {
  int idx = blockIdx.x * blockDim.x + threadIdx.x;
  if (idx >= BSZ * HSZ) return;
  int b = idx / HSZ, j = idx - b * HSZ;
  const float* gb = g + (size_t)b * 4 * HSZ;
  float iv = sigmoidf_(gb[j]);
  float fv = sigmoidf_(gb[HSZ + j]);
  float gv = tanhf(gb[2 * HSZ + j]);
  float ov = sigmoidf_(gb[3 * HSZ + j]);
  float cc = fv * c[idx] + iv * gv;
  c[idx] = cc;
  h[idx] = ov * tanhf(cc);
}

// ---------------------------------------------------------------------------
// Attention: scores = q . h_src ; softmax over S ; ctx = sum_s w_s * h_src.
// One block per batch element; LDS tree reductions.
// ---------------------------------------------------------------------------
__global__ void attn_kernel(const float* __restrict__ q,
                            const float* __restrict__ h_src,  // [S][B][H]
                            float* __restrict__ ctx) {
  __shared__ float red[256];
  __shared__ float w[SLEN];
  int b = blockIdx.x;
  int tid = threadIdx.x;
  const float* qb = q + (size_t)b * HSZ;
  for (int s = 0; s < SLEN; ++s) {
    const float* hs = h_src + ((size_t)s * BSZ + b) * HSZ;
    float p = 0.f;
    for (int j = tid; j < HSZ; j += 256) p += qb[j] * hs[j];
    red[tid] = p;
    __syncthreads();
    for (int off = 128; off > 0; off >>= 1) {
      if (tid < off) red[tid] += red[tid + off];
      __syncthreads();
    }
    if (tid == 0) w[s] = red[0];
    __syncthreads();
  }
  if (tid == 0) {
    float m = w[0];
    for (int s = 1; s < SLEN; ++s) m = fmaxf(m, w[s]);
    float sum = 0.f;
    for (int s = 0; s < SLEN; ++s) { float e = __expf(w[s] - m); w[s] = e; sum += e; }
    float inv = 1.f / sum;
    for (int s = 0; s < SLEN; ++s) w[s] *= inv;
  }
  __syncthreads();
  for (int j = tid; j < HSZ; j += 256) {
    float acc = 0.f;
    for (int s = 0; s < SLEN; ++s) acc += w[s] * h_src[((size_t)s * BSZ + b) * HSZ + j];
    ctx[(size_t)b * HSZ + j] = acc;
  }
}

// ---------------------------------------------------------------------------
// h_tilde = tanh([h_top || ctx] @ W_concat^T + b_concat); stored both to the
// recurrent input-feeding buffer and to h_tilde_seq[b][t][:].
// ---------------------------------------------------------------------------
__global__ void concat_proj_tanh_kernel(const float* __restrict__ h_top,
                                        const float* __restrict__ ctx,
                                        const float* __restrict__ Wc,  // [H][2H]
                                        const float* __restrict__ bc,
                                        float* __restrict__ h_tilde,
                                        float* __restrict__ hts,       // [B][T][H]
                                        int t) {
  int gwave = (blockIdx.x * blockDim.x + threadIdx.x) >> 5;
  int lane  = threadIdx.x & 31;
  if (gwave >= (HSZ >> 6)) return;
  int tg = gwave;
  const int ldw = 2 * HSZ;
  v8f acc[NB];
  zero_acc4(acc);
  wmma_tile4_f32f16(h_top, HSZ, Wc + (size_t)tg * 64 * ldw, ldw, HSZ, lane, acc);
  wmma_tile4_f32f16(ctx,  HSZ, Wc + (size_t)tg * 64 * ldw + HSZ, ldw, HSZ, lane, acc);
  int r0 = (lane >= 16) ? 8 : 0;
#pragma unroll
  for (int nb = 0; nb < NB; ++nb) {
    int col = tg * 64 + nb * 16 + (lane & 15);
#pragma unroll
    for (int i = 0; i < 8; ++i) {
      int b = r0 + i;
      float v = tanhf(acc[nb][i] + bc[col]);
      h_tilde[(size_t)b * HSZ + col] = v;
      hts[((size_t)b * TLEN + t) * HSZ + col] = v;
    }
  }
}

// ---------------------------------------------------------------------------
// In-place row-wise log-softmax over V columns (block per row).
// ---------------------------------------------------------------------------
__global__ void log_softmax_kernel(float* __restrict__ out, int V) {
  __shared__ float red[256];
  int row = blockIdx.x, tid = threadIdx.x;
  float* x = out + (size_t)row * V;
  float m = -1e30f;
  for (int j = tid; j < V; j += 256) m = fmaxf(m, x[j]);
  red[tid] = m;
  __syncthreads();
  for (int off = 128; off > 0; off >>= 1) {
    if (tid < off) red[tid] = fmaxf(red[tid], red[tid + off]);
    __syncthreads();
  }
  m = red[0];
  __syncthreads();
  float s = 0.f;
  for (int j = tid; j < V; j += 256) s += __expf(x[j] - m);
  red[tid] = s;
  __syncthreads();
  for (int off = 128; off > 0; off >>= 1) {
    if (tid < off) red[tid] += red[tid + off];
    __syncthreads();
  }
  float lse = m + __logf(red[0]);
  __syncthreads();
  for (int j = tid; j < V; j += 256) x[j] = x[j] - lse;
}

// ===========================================================================
extern "C" void kernel_launch(void* const* d_in, const int* in_sizes, int n_in,
                              void* d_out, int out_size, void* d_ws, size_t ws_size,
                              hipStream_t stream) {
  (void)in_sizes; (void)n_in; (void)out_size; (void)ws_size;
  const int*   src       = (const int*)d_in[0];
  const int*   tgt       = (const int*)d_in[1];
  const float* emb_src_W = (const float*)d_in[2];
  const float* emb_dec_W = (const float*)d_in[3];
  const float* enc_Wih   = (const float*)d_in[4];
  const float* enc_Whh   = (const float*)d_in[5];
  const float* enc_bih   = (const float*)d_in[6];
  const float* enc_bhh   = (const float*)d_in[7];
  const float* dec0_Wih  = (const float*)d_in[8];
  const float* dec0_Whh  = (const float*)d_in[9];
  const float* dec0_bih  = (const float*)d_in[10];
  const float* dec0_bhh  = (const float*)d_in[11];
  const float* dec_Wih   = (const float*)d_in[12];
  const float* dec_Whh   = (const float*)d_in[13];
  const float* dec_bih   = (const float*)d_in[14];
  const float* dec_bhh   = (const float*)d_in[15];
  const float* W_attn    = (const float*)d_in[16];
  const float* W_concat  = (const float*)d_in[17];
  const float* b_concat  = (const float*)d_in[18];
  const float* W_out     = (const float*)d_in[19];
  const float* b_out     = (const float*)d_in[20];
  float* out = (float*)d_out;

  // ---- carve workspace (fp32), ~28 MB ----
  float* ws = (float*)d_ws;
  size_t off = 0;
  auto alloc = [&](size_t n) { float* p = ws + off; off += (n + 63) & ~(size_t)63; return p; };
  float* bufA    = alloc((size_t)SLEN * BSZ * ESZ);          // encoder layer input
  float* bufB    = alloc((size_t)SLEN * BSZ * HSZ);          // encoder layer output
  float* xg      = alloc((size_t)2 * SLEN * BSZ * 4 * HDIR); // precomputed input proj
  float* hc_enc  = alloc((size_t)4 * BSZ * HDIR);            // h[2][B][HDIR] then c[2][B][HDIR]
  float* g_enc   = alloc((size_t)2 * BSZ * 4 * HDIR);
  float* dec_h   = alloc((size_t)LNUM * BSZ * HSZ);
  float* dec_c   = alloc((size_t)LNUM * BSZ * HSZ);
  float* emb_tgt = alloc((size_t)TLEN * BSZ * ESZ);
  float* g_dec   = alloc((size_t)BSZ * 4 * HSZ);
  float* h_tilde = alloc((size_t)BSZ * HSZ);
  float* qbuf    = alloc((size_t)BSZ * HSZ);
  float* ctx     = alloc((size_t)BSZ * HSZ);
  float* hts     = alloc((size_t)BSZ * TLEN * HSZ);          // h_tilde_seq [B][T][H]

  // ---- embeddings ----
  gather_embed_kernel<<<SLEN * BSZ, 256, 0, stream>>>(src, emb_src_W, bufA, BSZ, SLEN, ESZ);
  gather_embed_kernel<<<TLEN * BSZ, 256, 0, stream>>>(tgt, emb_dec_W, emb_tgt, BSZ, TLEN, ESZ);
  hipMemsetAsync(h_tilde, 0, (size_t)BSZ * HSZ * sizeof(float), stream);

  // ---- encoder: 4-layer bidirectional LSTM ----
  float* xin = bufA;
  float* xout = bufB;
  float* h_cur = hc_enc;
  float* c_cur = hc_enc + 2 * BSZ * HDIR;
  for (int l = 0; l < LNUM; ++l) {
    hipMemsetAsync(hc_enc, 0, (size_t)4 * BSZ * HDIR * sizeof(float), stream);
    // batched input projection over all timesteps (E == H == 1024, so K uniform)
    for (int d = 0; d < 2; ++d) {
      const float* Wih = enc_Wih + (size_t)(l * 2 + d) * (4 * HDIR) * ESZ;
      const float* bi  = enc_bih + (size_t)(l * 2 + d) * (4 * HDIR);
      const float* bh  = enc_bhh + (size_t)(l * 2 + d) * (4 * HDIR);
      int M = SLEN * BSZ, N = 4 * HDIR, K = ESZ;
      int waves = (M / 16) * (N / 64);   // 1600
      gemm_wmma_bias_kernel<<<(waves + 7) / 8, 256, 0, stream>>>(
          xin, K, Wih, K, bi, bh, xg + (size_t)d * SLEN * BSZ * N, N, M, N, K);
    }
    const float* Whh_l = enc_Whh + (size_t)l * 2 * (4 * HDIR) * HDIR;
    for (int st = 0; st < SLEN; ++st) {
      enc_step_gemm_kernel<<<dim3(4, 1, 2), 256, 0, stream>>>(h_cur, Whh_l, xg, g_enc, st);
      enc_cell_kernel<<<dim3(32, 1, 2), 256, 0, stream>>>(g_enc, h_cur, c_cur, xout, st,
                                                          dec_h, dec_c, l);
    }
    float* tmp = xin; xin = xout; xout = tmp;
  }
  float* h_src = xin;  // final encoder states, [S][B][H]

  // ---- decoder: teacher-forced, input feeding + attention ----
  for (int t = 0; t < TLEN; ++t) {
    // layer 0: x = [emb_t || h_tilde], K = 2048
    dec_gates_kernel<<<8, 256, 0, stream>>>(
        emb_tgt + (size_t)t * BSZ * ESZ, ESZ, h_tilde, HSZ,
        dec0_Wih, ESZ + HSZ, dec_h, dec0_Whh, dec0_bih, dec0_bhh, g_dec);
    dec_cell_kernel<<<64, 256, 0, stream>>>(g_dec, dec_h, dec_c);
    for (int i = 0; i < LNUM - 1; ++i) {
      float* h_below = dec_h + (size_t)i * BSZ * HSZ;
      float* h_l = dec_h + (size_t)(i + 1) * BSZ * HSZ;
      float* c_l = dec_c + (size_t)(i + 1) * BSZ * HSZ;
      dec_gates_kernel<<<8, 256, 0, stream>>>(
          h_below, HSZ, nullptr, 0,
          dec_Wih + (size_t)i * 4 * HSZ * HSZ, HSZ, h_l,
          dec_Whh + (size_t)i * 4 * HSZ * HSZ,
          dec_bih + (size_t)i * 4 * HSZ, dec_bhh + (size_t)i * 4 * HSZ, g_dec);
      dec_cell_kernel<<<64, 256, 0, stream>>>(g_dec, h_l, c_l);
    }
    float* h_top = dec_h + (size_t)(LNUM - 1) * BSZ * HSZ;
    {  // q = h_top @ W_attn^T
      int M = BSZ, N = HSZ, K = HSZ;
      int waves = (M / 16) * (N / 64);   // 16
      gemm_wmma_bias_kernel<<<(waves + 7) / 8, 256, 0, stream>>>(
          h_top, K, W_attn, K, nullptr, nullptr, qbuf, N, M, N, K);
    }
    attn_kernel<<<BSZ, 256, 0, stream>>>(qbuf, h_src, ctx);
    concat_proj_tanh_kernel<<<2, 256, 0, stream>>>(h_top, ctx, W_concat, b_concat,
                                                   h_tilde, hts, t);
  }

  // ---- generator: logits = hts @ W_out^T + b_out, then row log-softmax ----
  {
    int M = BSZ * TLEN, N = VOUTN, K = HSZ;
    int waves = (M / 16) * (N / 64);     // 50 * 500 = 25000
    gemm_wmma_bias_kernel<<<(waves + 7) / 8, 256, 0, stream>>>(
        hts, K, W_out, K, b_out, nullptr, out, N, M, N, K);
  }
  log_softmax_kernel<<<BSZ * TLEN, 256, 0, stream>>>(out, VOUTN);
}
